// RelationalGATLayer_3882650436802
// MI455X (gfx1250) — matrix operand order repaired and checked by reading, not compiled, not run on hardware
//
#include <hip/hip_runtime.h>
#include <hip/hip_bf16.h>

// ---------------------------------------------------------------------------
// Relational GAT layer for MI455X (gfx1250, wave32).
// GEMM via v_wmma_f32_16x16x32_bf16 with bf16-hi/lo error compensation
// (3 WMMA passes ~= fp32 accuracy). Edge phase is L2-bandwidth bound;
// everything fits in the 192MB L2.
// ---------------------------------------------------------------------------

typedef __bf16  v16bf __attribute__((ext_vector_type(16)));
typedef float   v8f   __attribute__((ext_vector_type(8)));
typedef unsigned int u32x4 __attribute__((ext_vector_type(4)));

union Frag32B { v16bf v; u32x4 q[2]; };

#define GAT_D 256
#define GAT_H 128
#define GAT_R 4

// -------------------------- init: out=0, m=NEG, sum=0 ----------------------
__global__ void __launch_bounds__(256)
gat_init(float* __restrict__ out, float* __restrict__ m, float* __restrict__ ssum,
         int nOut, int nRN) {
    int i = blockIdx.x * blockDim.x + threadIdx.x;
    if (i < nOut) out[i] = 0.0f;
    if (i < nRN) { m[i] = -1e30f; ssum[i] = 0.0f; }
}

// -------------------------- x -> bf16 hi/lo --------------------------------
__global__ void __launch_bounds__(256)
gat_cvt_x(const float* __restrict__ x, __bf16* __restrict__ xhi,
          __bf16* __restrict__ xlo, int n) {
    int i = blockIdx.x * blockDim.x + threadIdx.x;
    if (i >= n) return;
    float f  = x[i];
    __bf16 h = (__bf16)f;
    xhi[i] = h;
    xlo[i] = (__bf16)(f - (float)h);
}

// ---------------- W[r][k][n] -> Wt[r][n][k] bf16 hi/lo (B frags contiguous) -
__global__ void __launch_bounds__(256)
gat_cvt_w(const float* __restrict__ W, __bf16* __restrict__ whi,
          __bf16* __restrict__ wlo, int total) {
    int i = blockIdx.x * blockDim.x + threadIdx.x;
    if (i >= total) return;
    int r = i / (GAT_D * GAT_H);
    int k = (i / GAT_H) % GAT_D;
    int n = i % GAT_H;
    float f  = W[i];
    __bf16 h = (__bf16)f;
    size_t o = ((size_t)(r * GAT_H + n)) * GAT_D + k;
    whi[o] = h;
    wlo[o] = (__bf16)(f - (float)h);
}

// ------------------- WMMA GEMM: h[r] = x @ W[r] + b[r] ---------------------
// grid.x = row blocks of 128 (8 waves x 16 rows), grid.y = relation r.
// Each wave: 16x128 output strip, 8 f32 16x16 accumulators, A-frag reuse.
__global__ void __launch_bounds__(256)
gat_gemm_wmma(const __bf16* __restrict__ xhi, const __bf16* __restrict__ xlo,
              const __bf16* __restrict__ wthi, const __bf16* __restrict__ wtlo,
              const float* __restrict__ bias, float* __restrict__ h, int N) {
    const int lane = threadIdx.x & 31;
    const int wave = threadIdx.x >> 5;
    const int r    = blockIdx.y;
    const int rowBase = blockIdx.x * 128 + wave * 16;
    const int m16  = lane & 15;
    const int hi16 = lane >> 4;            // lane half selects K sub-range

    int arow = rowBase + m16;
    arow = (arow < N) ? arow : (N - 1);    // clamp: EXEC must stay all-ones
    const __bf16* Ah = xhi + (size_t)arow * GAT_D;
    const __bf16* Al = xlo + (size_t)arow * GAT_D;
    const __bf16* Bh = wthi + (size_t)r * GAT_H * GAT_D;
    const __bf16* Bl = wtlo + (size_t)r * GAT_H * GAT_D;

    v8f acc[8] = {};

    for (int k0 = 0; k0 < GAT_D; k0 += 32) {
        // A 16x32 bf16 frag: lanes 0-15 hold K 0-7 & 16-23, lanes 16-31 hold
        // K 8-15 & 24-31 (ISA 7.12.2). Two 16B loads per lane.
        const int ka = k0 + hi16 * 8;
        Frag32B fAh, fAl;
        fAh.q[0] = *(const u32x4*)(Ah + ka);
        fAh.q[1] = *(const u32x4*)(Ah + ka + 16);
        fAl.q[0] = *(const u32x4*)(Al + ka);
        fAl.q[1] = *(const u32x4*)(Al + ka + 16);

        const int kb = k0 + hi16 * 8;      // B: same K split as A (0-7 & 16-23 / 8-15 & 24-31)
        #pragma unroll
        for (int t = 0; t < 8; ++t) {
            const __bf16* bp = Bh + (size_t)(t * 16 + m16) * GAT_D + kb;
            const __bf16* bq = Bl + (size_t)(t * 16 + m16) * GAT_D + kb;
            Frag32B fBh, fBl;
            fBh.q[0] = *(const u32x4*)(bp);
            fBh.q[1] = *(const u32x4*)(bp + 16);
            fBl.q[0] = *(const u32x4*)(bq);
            fBl.q[1] = *(const u32x4*)(bq + 16);
            // bf16x3 compensated product: Ah*Bh + Ah*Bl + Al*Bh
            acc[t] = __builtin_amdgcn_wmma_f32_16x16x32_bf16(
                false, fAh.v, false, fBh.v, (short)0, acc[t], false, false);
            acc[t] = __builtin_amdgcn_wmma_f32_16x16x32_bf16(
                false, fAh.v, false, fBl.v, (short)0, acc[t], false, false);
            acc[t] = __builtin_amdgcn_wmma_f32_16x16x32_bf16(
                false, fAl.v, false, fBh.v, (short)0, acc[t], false, false);
        }
    }

    // C layout: VGPR v -> M = v + 8*hi16, N = lane&15 (per 16x16 tile).
    const float* brow = bias + r * GAT_H;
    float* hout = h + (size_t)r * N * GAT_H;
    #pragma unroll
    for (int t = 0; t < 8; ++t) {
        int col  = t * 16 + m16;
        float bv = brow[col];
        #pragma unroll
        for (int v = 0; v < 8; ++v) {
            int row = rowBase + v + hi16 * 8;
            if (row < N) hout[(size_t)row * GAT_H + col] = acc[t][v] + bv;
        }
    }
}

// ----------------- per-node attention dots: s1 = h.a1, s2 = h.a2 -----------
__global__ void __launch_bounds__(256)
gat_svec(const float* __restrict__ h, const float* __restrict__ a,
         float* __restrict__ s1, float* __restrict__ s2, int totalRN) {
    int lane = threadIdx.x & 31;
    int wv   = (blockIdx.x * blockDim.x + threadIdx.x) >> 5;
    if (wv >= totalRN) return;
    const float4* hv = (const float4*)(h + (size_t)wv * GAT_H);
    const float4* a1 = (const float4*)(a);
    const float4* a2 = (const float4*)(a + GAT_H);
    float4 xv = hv[lane], p = a1[lane], q = a2[lane];
    float d1 = xv.x * p.x + xv.y * p.y + xv.z * p.z + xv.w * p.w;
    float d2 = xv.x * q.x + xv.y * q.y + xv.z * q.z + xv.w * q.w;
    #pragma unroll
    for (int off = 16; off > 0; off >>= 1) {
        d1 += __shfl_xor(d1, off, 32);
        d2 += __shfl_xor(d2, off, 32);
    }
    if (lane == 0) { s1[wv] = d1; s2[wv] = d2; }
}

// ------------- float atomic max via sign-split int/uint atomics ------------
__device__ __forceinline__ void atomicMaxF(float* addr, float v) {
    if (v >= 0.0f) atomicMax((int*)addr, __float_as_int(v));
    else           atomicMin((unsigned int*)addr, __float_as_uint(v));
}

// ------------------------- edge pass 1: segment max ------------------------
__global__ void __launch_bounds__(256)
gat_edge_max(const int* __restrict__ ei, const int* __restrict__ et,
             const float* __restrict__ s1, const float* __restrict__ s2,
             float* __restrict__ sbuf, float* __restrict__ m, int E, int N) {
    int e = blockIdx.x * blockDim.x + threadIdx.x;
    if (e >= E) return;
    int r = et[e], row = ei[e], col = ei[E + e];
    size_t rn = (size_t)r * N;
    float sc = s1[rn + row] + s2[rn + col];
    sc = (sc >= 0.0f) ? sc : 0.2f * sc;          // leaky_relu, slope 0.2
    sbuf[e] = sc;
    atomicMaxF(m + rn + row, sc);
}

// --------------------- edge pass 2: exp & segment sum ----------------------
__global__ void __launch_bounds__(256)
gat_edge_sum(const int* __restrict__ ei, const int* __restrict__ et,
             const float* __restrict__ m, float* __restrict__ sbuf,
             float* __restrict__ ssum, int E, int N) {
    int e = blockIdx.x * blockDim.x + threadIdx.x;
    if (e >= E) return;
    int r = et[e], row = ei[e];
    size_t rn = (size_t)r * N;
    float w = __expf(sbuf[e] - m[rn + row]);
    sbuf[e] = w;
    atomicAdd(ssum + rn + row, w);
}

// ------------- edge pass 3: out[row] += alpha * h[r][col] (wave/edge) ------
__global__ void __launch_bounds__(256)
gat_scatter(const int* __restrict__ ei, const int* __restrict__ et,
            const float* __restrict__ sbuf, const float* __restrict__ ssum,
            const float* __restrict__ h, float* __restrict__ out, int E, int N) {
    int lane = threadIdx.x & 31;
    int e    = (blockIdx.x * blockDim.x + threadIdx.x) >> 5;
    if (e >= E) return;
    int r = et[e], row = ei[e], col = ei[E + e];
    size_t rn = (size_t)r * N;
    float sv    = ssum[rn + row];
    float alpha = sbuf[e] / ((sv > 0.0f) ? sv : 1.0f);
    const float4* hc = (const float4*)(h + (rn + col) * GAT_H);
    float4 hv = hc[lane];                         // 32 lanes x 16B = 128 floats
    float* op = out + (size_t)row * GAT_H + lane * 4;
    atomicAdd(op + 0, hv.x * alpha);
    atomicAdd(op + 1, hv.y * alpha);
    atomicAdd(op + 2, hv.z * alpha);
    atomicAdd(op + 3, hv.w * alpha);
}

// ---------------------------------------------------------------------------
extern "C" void kernel_launch(void* const* d_in, const int* in_sizes, int n_in,
                              void* d_out, int out_size, void* d_ws, size_t ws_size,
                              hipStream_t stream) {
    const float* x  = (const float*)d_in[0];
    const int*   ei = (const int*)d_in[1];     // (2,E) int32 (JAX x64 off)
    const int*   et = (const int*)d_in[2];     // (E,)  int32
    const float* a  = (const float*)d_in[3];   // (1, 2H, 1)
    const float* W  = (const float*)d_in[4];   // (R, D, H)
    const float* b  = (const float*)d_in[5];   // (R, H)
    float* out = (float*)d_out;

    const int N = in_sizes[0] / GAT_D;         // 50000
    const int E = in_sizes[2];                 // 640000
    const int RN = GAT_R * N;

    // ---- workspace carve-up (256B aligned) ----
    char* ws = (char*)d_ws;
    size_t off = 0;
    auto carve = [&](size_t bytes) -> void* {
        off = (off + 255) & ~(size_t)255;
        void* p = ws + off;
        off += bytes;
        return p;
    };
    __bf16* xhi  = (__bf16*)carve((size_t)N * GAT_D * 2);
    __bf16* xlo  = (__bf16*)carve((size_t)N * GAT_D * 2);
    __bf16* wthi = (__bf16*)carve((size_t)GAT_R * GAT_H * GAT_D * 2);
    __bf16* wtlo = (__bf16*)carve((size_t)GAT_R * GAT_H * GAT_D * 2);
    float*  hbuf = (float*)carve((size_t)GAT_R * N * GAT_H * 4);
    float*  s1   = (float*)carve((size_t)RN * 4);
    float*  s2   = (float*)carve((size_t)RN * 4);
    float*  mseg = (float*)carve((size_t)RN * 4);
    float*  ssum = (float*)carve((size_t)RN * 4);
    float*  sbuf = (float*)carve((size_t)E * 4);
    (void)ws_size; (void)n_in; (void)out_size;

    const int nOut = N * GAT_H;

    gat_init<<<(nOut + 255) / 256, 256, 0, stream>>>(out, mseg, ssum, nOut, RN);

    gat_cvt_x<<<(N * GAT_D + 255) / 256, 256, 0, stream>>>(x, xhi, xlo, N * GAT_D);

    const int wTot = GAT_R * GAT_D * GAT_H;
    gat_cvt_w<<<(wTot + 255) / 256, 256, 0, stream>>>(W, wthi, wtlo, wTot);

    dim3 ggrid((N + 127) / 128, GAT_R);
    gat_gemm_wmma<<<ggrid, 256, 0, stream>>>(xhi, xlo, wthi, wtlo, b, hbuf, N);

    gat_svec<<<(RN * 32 + 255) / 256, 256, 0, stream>>>(hbuf, a, s1, s2, RN);

    gat_edge_max<<<(E + 255) / 256, 256, 0, stream>>>(ei, et, s1, s2, sbuf, mseg, E, N);

    gat_edge_sum<<<(E + 255) / 256, 256, 0, stream>>>(ei, et, mseg, sbuf, ssum, E, N);

    gat_scatter<<<((size_t)E * 32 + 255) / 256, 256, 0, stream>>>(ei, et, sbuf, ssum,
                                                                  hbuf, out, E, N);
}
